// _NodeConv_72834055406396
// MI455X (gfx1250) — compile-verified, matrix operand order checked by reading, not run on hardware
//
#include <hip/hip_runtime.h>
#include <stdint.h>

#define N_NODES   100000
#define IN_F      32
#define OUT_F     32
#define EDGE_F    16
#define N_EDGES   400000
#define K_PROD    (EDGE_F * IN_F)      // 512
#define K_TOT     (K_PROD + IN_F)      // 544 (bias term folded in via x row)
#define N_KC      (K_TOT / 32)         // 17 k-chunks of 32
#define BN_EPS    1e-5f
#define SLOPE     0.01f

typedef __attribute__((ext_vector_type(16))) __bf16         v16bf;
typedef __attribute__((ext_vector_type(16))) unsigned short v16u;
typedef __attribute__((ext_vector_type(16))) float          v16f;
typedef __attribute__((ext_vector_type(8)))  float          v8f;

// ISA 16-bit A-matrix 16x32 layout (CDNA5 ISA):
// lane hi-half=0: element e -> K = e (e<8) / e+8 (e>=8)   => K in {0..7,16..23}
// lane hi-half=1: same +8                                  => K in {8..15,24..31}
__device__ __forceinline__ int kmap(int e, int hi) {
  return e + (e & 8) + hi * 8;
}

// float -> bf16 bits, round-to-nearest-even (cold path / prep only)
__device__ __forceinline__ unsigned short f2bf(float f) {
  unsigned int u = __float_as_uint(f);
  u += 0x7FFFu + ((u >> 16) & 1u);
  return (unsigned short)(u >> 16);
}

// ---------------------------------------------------------------------------
// prep: zero BN stats; build bf16 B-fragments for W' = [nn_w reshaped ; nn_b
// reshaped] (544x32) and root (32x32), pre-swizzled so a wave loads a
// 16x(32-k-chunk) B fragment with two global_load_b128.
// wfrag index = ((kc*2 + ntile)*32 + lane)*16 + e
// ---------------------------------------------------------------------------
__global__ __launch_bounds__(256) void prep_kernel(
    const float* __restrict__ nn_w, const float* __restrict__ nn_b,
    const float* __restrict__ root, float* __restrict__ stats,
    unsigned short* __restrict__ wfrag, unsigned short* __restrict__ rfrag) {
  const int tid = threadIdx.x;
  if (tid < 2 * OUT_F) stats[tid] = 0.0f;

  for (int idx = tid; idx < N_KC * 2 * 32 * 16; idx += 256) {
    int e    = idx & 15;
    int lane = (idx >> 4) & 31;
    int nt   = (idx >> 9) & 1;
    int kc   = idx >> 10;
    int hi   = lane >> 4;
    int n    = nt * 16 + (lane & 15);
    int k    = kc * 32 + kmap(e, hi);
    float v;
    if (k < K_PROD) {                      // k = f*32 + i
      int f = k >> 5, i = k & 31;
      v = nn_w[f * (IN_F * OUT_F) + i * OUT_F + n];
    } else {                               // bias part: row i of nn_b matrix
      int i = k - K_PROD;
      v = nn_b[i * OUT_F + n];
    }
    wfrag[idx] = f2bf(v);
  }

  for (int idx = tid; idx < 2 * 32 * 16; idx += 256) {
    int e    = idx & 15;
    int lane = (idx >> 4) & 31;
    int nt   = idx >> 9;
    int hi   = lane >> 4;
    int n    = nt * 16 + (lane & 15);
    int k    = kmap(e, hi);                // K dim = IN_F = 32, one chunk
    rfrag[idx] = f2bf(root[k * OUT_F + n]);
  }
}

// ---------------------------------------------------------------------------
// root: h = X @ root + bias  (fully overwrites d_out each call)
// One wave per 16-node tile; 1 k-chunk, 2 n-tiles => 2 WMMAs.
// ---------------------------------------------------------------------------
__global__ __launch_bounds__(256) void root_kernel(
    const float* __restrict__ x, const float* __restrict__ bias,
    const unsigned short* __restrict__ rfrag, float* __restrict__ out) {
  const int lane = threadIdx.x & 31;
  const int wave = threadIdx.x >> 5;
  const int tile = blockIdx.x * 8 + wave;          // 16 nodes per tile
  if (tile * 16 >= N_NODES) return;                // uniform per wave
  const int hi = lane >> 4;
  const int m0 = lane & 15;

  const float* xr = x + (size_t)(tile * 16 + m0) * IN_F;
  v16f xvv;
#pragma unroll
  for (int e = 0; e < 16; ++e) xvv[e] = xr[kmap(e, hi)];
  v16bf a = __builtin_convertvector(xvv, v16bf);   // native packed f32->bf16

  const v16u* bf = (const v16u*)rfrag;
  v16bf b0 = __builtin_bit_cast(v16bf, bf[0 * 32 + lane]);
  v16bf b1 = __builtin_bit_cast(v16bf, bf[1 * 32 + lane]);

  v8f acc0 = {}, acc1 = {};
  acc0 = __builtin_amdgcn_wmma_f32_16x16x32_bf16(false, a, false, b0, (short)0, acc0, false, false);
  acc1 = __builtin_amdgcn_wmma_f32_16x16x32_bf16(false, a, false, b1, (short)0, acc1, false, false);

  const float bn0 = bias[m0];
  const float bn1 = bias[16 + m0];
#pragma unroll
  for (int r = 0; r < 8; ++r) {                    // D: lane holds col n, rows r+8*hi
    int nd = tile * 16 + r + 8 * hi;
    out[(size_t)nd * OUT_F + m0]      = acc0[r] + bn0;
    out[(size_t)nd * OUT_F + 16 + m0] = acc1[r] + bn1;
  }
}

// ---------------------------------------------------------------------------
// edge: msg = Z' @ W' as WMMA GEMM, scatter-add into d_out via fp32 atomics.
// Z'[e, f*32+i] = ef[e,f] * x[src[e],i];  Z'[e, 512+i] = x[src[e],i]
// One wave per 16-edge tile: 17 k-chunks x 2 n-tiles = 34 WMMAs.
// A fragment per chunk = (vector splat-mul) + (packed f32->bf16 convert).
// ---------------------------------------------------------------------------
__global__ __launch_bounds__(256) void edge_kernel(
    const float* __restrict__ x, const float* __restrict__ ef,
    const long long* __restrict__ eidx, const unsigned short* __restrict__ wfrag,
    float* __restrict__ out) {
  const int lane = threadIdx.x & 31;
  const int wave = threadIdx.x >> 5;
  const int tile = blockIdx.x * 8 + wave;          // 25000 tiles exactly
  const int hi = lane >> 4;
  const int m0 = lane & 15;

  const long long edge = (long long)tile * 16 + m0;
  const long long s = eidx[edge];                  // src node id (int64)

  // Per-lane row data in registers (b128-friendly K pattern within the row).
  const float* xr = x + (size_t)s * IN_F;
  v16f xvv;
#pragma unroll
  for (int e = 0; e < 16; ++e) xvv[e] = xr[kmap(e, hi)];
  float efv[EDGE_F];
  const float* er = ef + (size_t)edge * EDGE_F;
#pragma unroll
  for (int f = 0; f < EDGE_F; ++f) efv[f] = er[f];

  const v16u* bf = (const v16u*)wfrag;
  v8f acc0 = {}, acc1 = {};
#pragma unroll
  for (int kc = 0; kc < N_KC; ++kc) {
    v16bf a;
    if (kc < 16) {
      v16f p = xvv * efv[kc];                      // v_pk_mul_f32 / dual-issue
      a = __builtin_convertvector(p, v16bf);       // packed cvt f32->bf16
    } else {                                       // bias chunk: raw x row
      a = __builtin_convertvector(xvv, v16bf);
    }
    v16bf b0 = __builtin_bit_cast(v16bf, bf[(kc * 2 + 0) * 32 + lane]);
    v16bf b1 = __builtin_bit_cast(v16bf, bf[(kc * 2 + 1) * 32 + lane]);
    acc0 = __builtin_amdgcn_wmma_f32_16x16x32_bf16(false, a, false, b0, (short)0, acc0, false, false);
    acc1 = __builtin_amdgcn_wmma_f32_16x16x32_bf16(false, a, false, b1, (short)0, acc1, false, false);
  }

  // Scatter-add: D-layout row m = r + 8*hi, col n = m0 (+16 for tile 1).
  const long long* dsts = eidx + N_EDGES;
  const long long base = (long long)tile * 16;
#pragma unroll
  for (int r = 0; r < 8; ++r) {
    long long d = dsts[base + r + 8 * hi];
    float* p = out + (size_t)d * OUT_F + m0;
    __hip_atomic_fetch_add(p,      acc0[r], __ATOMIC_RELAXED, __HIP_MEMORY_SCOPE_AGENT);
    __hip_atomic_fetch_add(p + 16, acc1[r], __ATOMIC_RELAXED, __HIP_MEMORY_SCOPE_AGENT);
  }
}

// ---------------------------------------------------------------------------
// stats: per-feature sum / sum-of-squares (LDS ds_add_f32 + global atomics)
// ---------------------------------------------------------------------------
__global__ __launch_bounds__(256) void stats_kernel(
    const float* __restrict__ h, float* __restrict__ stats) {
  __shared__ float ssum[OUT_F];
  __shared__ float ssq[OUT_F];
  const int tid = threadIdx.x;
  if (tid < OUT_F) { ssum[tid] = 0.0f; ssq[tid] = 0.0f; }
  __syncthreads();

  const int f = tid & 31;                          // stride keeps feature fixed
  float ps = 0.0f, pq = 0.0f;
  const size_t total = (size_t)N_NODES * OUT_F;
  for (size_t i = (size_t)blockIdx.x * blockDim.x + tid; i < total;
       i += (size_t)gridDim.x * blockDim.x) {
    float v = h[i];
    ps += v;
    pq += v * v;
  }
  __hip_atomic_fetch_add(&ssum[f], ps, __ATOMIC_RELAXED, __HIP_MEMORY_SCOPE_WORKGROUP);
  __hip_atomic_fetch_add(&ssq[f],  pq, __ATOMIC_RELAXED, __HIP_MEMORY_SCOPE_WORKGROUP);
  __syncthreads();
  if (tid < OUT_F) {
    __hip_atomic_fetch_add(&stats[tid],         ssum[tid], __ATOMIC_RELAXED, __HIP_MEMORY_SCOPE_AGENT);
    __hip_atomic_fetch_add(&stats[OUT_F + tid], ssq[tid],  __ATOMIC_RELAXED, __HIP_MEMORY_SCOPE_AGENT);
  }
}

// ---------------------------------------------------------------------------
// norm: BatchNorm (biased var) + LeakyReLU, in place on d_out
// ---------------------------------------------------------------------------
__global__ __launch_bounds__(256) void norm_kernel(
    const float* __restrict__ stats, const float* __restrict__ gamma,
    const float* __restrict__ beta, float* __restrict__ out) {
  const size_t i = (size_t)blockIdx.x * blockDim.x + threadIdx.x;
  if (i >= (size_t)N_NODES * OUT_F) return;
  const int f = (int)(i & 31);
  const float inv_n = 1.0f / (float)N_NODES;
  const float mean = stats[f] * inv_n;
  float var = stats[OUT_F + f] * inv_n - mean * mean;
  var = var < 0.0f ? 0.0f : var;
  const float is = rsqrtf(var + BN_EPS);
  float v = (out[i] - mean) * is * gamma[f] + beta[f];
  out[i] = v >= 0.0f ? v : SLOPE * v;
}

extern "C" void kernel_launch(void* const* d_in, const int* in_sizes, int n_in,
                              void* d_out, int out_size, void* d_ws, size_t ws_size,
                              hipStream_t stream) {
  (void)in_sizes; (void)n_in; (void)out_size; (void)ws_size;
  const float*     node_feature = (const float*)d_in[0];
  const float*     edge_feature = (const float*)d_in[1];
  const float*     nn_w         = (const float*)d_in[2];
  const float*     nn_b         = (const float*)d_in[3];
  const float*     root         = (const float*)d_in[4];
  const float*     bias         = (const float*)d_in[5];
  const float*     gamma        = (const float*)d_in[6];
  const float*     beta         = (const float*)d_in[7];
  const long long* eidx         = (const long long*)d_in[8];
  float* out = (float*)d_out;

  // workspace layout: [0,256) stats; then W' fragments; then root fragments
  char* ws = (char*)d_ws;
  float*          stats = (float*)ws;
  unsigned short* wfrag = (unsigned short*)(ws + 256);
  unsigned short* rfrag = (unsigned short*)(ws + 256 + (size_t)N_KC * 2 * 32 * 16 * 2);

  prep_kernel<<<1, 256, 0, stream>>>(nn_w, nn_b, root, stats, wfrag, rfrag);
  root_kernel<<<(N_NODES / 16 + 7) / 8, 256, 0, stream>>>(node_feature, bias, rfrag, out);
  edge_kernel<<<N_EDGES / (16 * 8), 256, 0, stream>>>(node_feature, edge_feature, eidx, wfrag, out);
  stats_kernel<<<512, 256, 0, stream>>>(out, stats);
  norm_kernel<<<((size_t)N_NODES * OUT_F + 255) / 256, 256, 0, stream>>>(stats, gamma, beta, out);
}
